// Resnet18_41635412967668
// MI455X (gfx1250) — matrix operand order repaired and checked by reading, not compiled, not run on hardware
//
#include <hip/hip_runtime.h>
#include <stdint.h>

// ---------------------------------------------------------------------------
// CDNA5 (gfx1250) ResNet-18 forward: bf16 activations/weights, f32 accumulate
// via v_wmma_f32_16x16x32_bf16. Wave32 implicit-GEMM convolutions with
// async-to-LDS weight staging (ASYNCcnt) when the toolchain exposes it.
// ---------------------------------------------------------------------------

typedef __attribute__((ext_vector_type(16))) __bf16        v16bf;
typedef __attribute__((ext_vector_type(8)))  float         v8f;
typedef __attribute__((ext_vector_type(16))) unsigned short v16u;

// Generic-vector int4: this (not HIP's struct int4, not ext_vector) is the
// pointee type of __builtin_amdgcn_global_load_async_to_lds_b128's params.
typedef int vi4 __attribute__((vector_size(16)));
typedef __attribute__((address_space(1))) vi4* gvi4p;   // global (== __device__)
typedef __attribute__((address_space(3))) vi4* lvi4p;   // LDS    (== __shared__)

#if defined(__HIP_DEVICE_COMPILE__) && \
    __has_builtin(__builtin_amdgcn_global_load_async_to_lds_b128) && \
    __has_builtin(__builtin_amdgcn_s_wait_asynccnt)
#define HAVE_ASYNC_LDS 1
#else
#define HAVE_ASYNC_LDS 0
#endif

__device__ __forceinline__ float bf2f(unsigned short u) {
    unsigned int x = ((unsigned int)u) << 16;
    return __builtin_bit_cast(float, x);
}
__device__ __forceinline__ unsigned short f2bf(float f) {
    unsigned int x = __builtin_bit_cast(unsigned int, f);
    unsigned int r = x + 0x7FFFu + ((x >> 16) & 1u);   // round-to-nearest-even
    return (unsigned short)(r >> 16);
}

// ---------------------------------------------------------------------------
// fp32 -> bf16 elementwise pack (for the input image)
// ---------------------------------------------------------------------------
__global__ void pack_f32_bf16(const float* __restrict__ src,
                              unsigned short* __restrict__ dst, int n) {
    int i = blockIdx.x * 256 + threadIdx.x;
    if (i < n) dst[i] = f2bf(src[i]);
}

// ---------------------------------------------------------------------------
// Weight pack: fp32 OIHW [Cout][K] -> bf16 [Cout][Kpad], zero padded K tail.
// ---------------------------------------------------------------------------
__global__ void pack_weight_bf16(const float* __restrict__ src,
                                 unsigned short* __restrict__ dst,
                                 int Cout, int K, int Kpad) {
    int i = blockIdx.x * 256 + threadIdx.x;
    int total = Cout * Kpad;
    if (i >= total) return;
    int co = i / Kpad;
    int k  = i - co * Kpad;
    float v = (k < K) ? src[(size_t)co * K + k] : 0.0f;
    dst[i] = f2bf(v);
}

// ---------------------------------------------------------------------------
// Per-lane output-pixel descriptor
// ---------------------------------------------------------------------------
struct Pix { int nn, ho, wo, ih0, iw0; bool valid; };

__device__ __forceinline__ Pix decode_pix(int p, int totalPix, int Hout,
                                          int Wout, int stride, int pad) {
    Pix px;
    px.valid = (p < totalPix);
    const int pc = px.valid ? p : 0;
    px.wo = pc % Wout;
    const int t = pc / Wout;
    px.ho = t % Hout;
    px.nn = t / Hout;
    px.ih0 = px.ho * stride - pad;
    px.iw0 = px.wo * stride - pad;
    return px;
}

// im2col gather of K elements kb+0..7 and kb+16..23 (B-fragment half-column).
// TKH/TKW compile-time -> constant-divisor strength reduction.
template <int TKH, int TKW>
__device__ __forceinline__ v16u gather_b(const unsigned short* __restrict__ in,
                                         const Pix& px, int kb, int K,
                                         int Cin, int Hin, int Win) {
    constexpr int KHW = TKH * TKW;
    v16u bu = {};
#pragma unroll
    for (int i = 0; i < 16; ++i) {
        const int k = kb + (i & 7) + ((i >> 3) << 4);
        unsigned short v = 0;
        if (px.valid && k < K) {
            const int ci = k / KHW;
            const int r  = k - ci * KHW;
            const int kh = r / TKW;
            const int kw = r - kh * TKW;
            const int ih = px.ih0 + kh;
            const int iw = px.iw0 + kw;
            if (ih >= 0 && ih < Hin && iw >= 0 && iw < Win)
                v = in[(((size_t)px.nn * Cin + ci) * Hin + ih) * Win + iw];
        }
        bu[i] = v;
    }
    return bu;
}

__device__ __forceinline__ void store_tile(const v8f& acc, const Pix& px,
                                           int coBase, int half,
                                           const float* __restrict__ scale,
                                           const float* __restrict__ bias,
                                           const unsigned short* __restrict__ residual,
                                           unsigned short* __restrict__ out_bf,
                                           float* __restrict__ out_f32,
                                           int Cout, int Hout, int Wout,
                                           int do_relu) {
    if (!px.valid) return;
#pragma unroll
    for (int r = 0; r < 8; ++r) {
        const int co = coBase + half * 8 + r;
        if (co < Cout) {
            float v = acc[r];
            v = v * (scale ? scale[co] : 1.0f) + (bias ? bias[co] : 0.0f);
            const size_t oidx =
                (((size_t)px.nn * Cout + co) * Hout + px.ho) * Wout + px.wo;
            if (residual) v += bf2f(residual[oidx]);
            if (do_relu)  v = (v > 0.0f) ? v : 0.0f;
            if (out_f32)  out_f32[oidx] = v;
            else          out_bf[oidx]  = f2bf(v);
        }
    }
}

// ---------------------------------------------------------------------------
// Implicit-GEMM conv + BN(scale,bias) + optional residual + optional ReLU.
// One wave32 computes a 16(Cout) x 32(B*Ho*Wo) output tile:
//   - A (weights, 16x32 bf16) staged per wave into LDS by
//     GLOBAL_LOAD_ASYNC_TO_LDS_B128 (ASYNCcnt), read back as ds_load_b128,
//     reused by two v_wmma_f32_16x16x32_bf16 per K-step.
//   - B fragments gathered im2col-style per lane.
//
// Fragment layouts (CDNA5 ISA 7.12.2, wave32):
//   A (16x32, 16-bit): lane M = lane&15; K = half*8 + {0..7} and half*8 + {16..23}
//   B (32x16, 16-bit): lane N = lane&15; K = half*8 + {0..7} and half*8 + {16..23}
//   C (16x16, f32)   : lane N = lane&15; VGPR r holds M = half*8 + r
// ---------------------------------------------------------------------------
template <int TKH, int TKW>
__global__ __launch_bounds__(128)
void conv_bn_wmma(const unsigned short* __restrict__ in,
                  const unsigned short* __restrict__ wt,
                  const float* __restrict__ scale,
                  const float* __restrict__ bias,
                  const unsigned short* __restrict__ residual,
                  unsigned short* __restrict__ out_bf,
                  float* __restrict__ out_f32,
                  int B, int Cin, int Hin, int Win,
                  int Cout, int Hout, int Wout,
                  int stride, int pad, int Kpad, int do_relu)
{
    constexpr int KHW = TKH * TKW;
    const int K        = Cin * KHW;
    const int totalPix = B * Hout * Wout;
    const int pixTiles = (totalPix + 31) >> 5;
    const int coTiles  = (Cout + 15) >> 4;
    const int tiles    = pixTiles * coTiles;

    const int wid  = threadIdx.x >> 5;          // wave id (wave32)
    const int tile = blockIdx.x * 4 + wid;
    if (tile >= tiles) return;                  // wave-uniform exit

    const int cTile = tile / pixTiles;
    const int pTile = tile - cTile * pixTiles;

    const int lane  = threadIdx.x & 31;
    const int laneM = lane & 15;
    const int half  = lane >> 4;
    const int coBase = cTile * 16;
    const int pBase  = pTile * 32;

    const Pix px0 = decode_pix(pBase + laneM,      totalPix, Hout, Wout, stride, pad);
    const Pix px1 = decode_pix(pBase + 16 + laneM, totalPix, Hout, Wout, stride, pad);

    // A-matrix weight row for this lane's fragment
    const int coA   = coBase + laneM;
    const int coClp = (coA < Cout) ? coA : (Cout - 1);
    const unsigned short* wrow = wt + (size_t)coClp * Kpad + half * 8;

#if HAVE_ASYNC_LDS
    // Per-wave 16x32 bf16 A tile in LDS, filled by async DMA each K-step.
    __shared__ __align__(16) unsigned short ldsA[4][16 * 32];
    unsigned short* tileA = ldsA[wid];
    // 64 16B chunks cover the tile; each lane copies chunk lane and lane+32.
    const int r0  = lane >> 2;                  // rows 0..7
    const int chk = lane & 3;                   // 4 chunks per 32-elem row
    const int gco0 = (coBase + r0     < Cout) ? (coBase + r0)     : (Cout - 1);
    const int gco1 = (coBase + r0 + 8 < Cout) ? (coBase + r0 + 8) : (Cout - 1);
    const unsigned short* gsrc0 = wt + (size_t)gco0 * Kpad + chk * 8;
    const unsigned short* gsrc1 = wt + (size_t)gco1 * Kpad + chk * 8;
    unsigned short* ldst0 = tileA + (r0)     * 32 + chk * 8;
    unsigned short* ldst1 = tileA + (r0 + 8) * 32 + chk * 8;
#endif

    const v16u vzero = {};
    v8f acc0 = {};
    v8f acc1 = {};

    for (int k0 = 0; k0 < Kpad; k0 += 32) {
        // ---- stage + read A fragment ----
        v16u au;
#if HAVE_ASYNC_LDS
        __builtin_amdgcn_global_load_async_to_lds_b128(
            (gvi4p)(gsrc0 + k0), (lvi4p)ldst0, 0, 0);
        __builtin_amdgcn_global_load_async_to_lds_b128(
            (gvi4p)(gsrc1 + k0), (lvi4p)ldst1, 0, 0);
        __builtin_amdgcn_s_wait_asynccnt(0);
        ((uint4*)&au)[0] = *(const uint4*)(tileA + laneM * 32 + half * 8);
        ((uint4*)&au)[1] = *(const uint4*)(tileA + laneM * 32 + half * 8 + 16);
#else
        ((uint4*)&au)[0] = *(const uint4*)(wrow + k0);       // K = half*8 + 0..7
        ((uint4*)&au)[1] = *(const uint4*)(wrow + k0 + 16);  // K = half*8 + 16..23
#endif
        if (coA >= Cout) au = vzero;

        // Prefetch next weight slice (global_prefetch_b8)
        if (k0 + 32 < Kpad) __builtin_prefetch(wrow + k0 + 32, 0, 0);

        // ---- B fragments for the two pixel half-tiles ----
        const int kb = k0 + half * 8;
        const v16u bu0 = gather_b<TKH, TKW>(in, px0, kb, K, Cin, Hin, Win);
        const v16u bu1 = gather_b<TKH, TKW>(in, px1, kb, K, Cin, Hin, Win);

        const v16bf a  = __builtin_bit_cast(v16bf, au);
        const v16bf b0 = __builtin_bit_cast(v16bf, bu0);
        const v16bf b1 = __builtin_bit_cast(v16bf, bu1);
        acc0 = __builtin_amdgcn_wmma_f32_16x16x32_bf16(
                   false, a, false, b0, (short)0, acc0, false, false);
        acc1 = __builtin_amdgcn_wmma_f32_16x16x32_bf16(
                   false, a, false, b1, (short)0, acc1, false, false);
    }

    // ---- Epilogue: BN affine + residual + ReLU, bf16 or f32 store ----
    store_tile(acc0, px0, coBase, half, scale, bias, residual,
               out_bf, out_f32, Cout, Hout, Wout, do_relu);
    store_tile(acc1, px1, coBase, half, scale, bias, residual,
               out_bf, out_f32, Cout, Hout, Wout, do_relu);
}

// ---------------------------------------------------------------------------
// _cut: zero a (b,c) plane whose nonzero fraction < 0.2.
// ---------------------------------------------------------------------------
__global__ void cut_kernel(unsigned short* __restrict__ act, int planeSize) {
    __shared__ int cnt[256];
    const size_t base = (size_t)blockIdx.x * planeSize;
    int c = 0;
    for (int i = threadIdx.x; i < planeSize; i += 256)
        c += ((act[base + i] & 0x7FFFu) != 0);
    cnt[threadIdx.x] = c;
    __syncthreads();
    for (int s = 128; s > 0; s >>= 1) {
        if ((int)threadIdx.x < s) cnt[threadIdx.x] += cnt[threadIdx.x + s];
        __syncthreads();
    }
    const float frac = (float)cnt[0] / (float)planeSize;
    if (frac < 0.2f) {
        for (int i = threadIdx.x; i < planeSize; i += 256)
            act[base + i] = 0;
    }
}

// ---------------------------------------------------------------------------
// 3x3 stride-2 pad-1 max pool (bf16)
// ---------------------------------------------------------------------------
__global__ void maxpool_kernel(const unsigned short* __restrict__ in,
                               unsigned short* __restrict__ out,
                               int BC, int Hin, int Win, int Hout, int Wout) {
    const int total = BC * Hout * Wout;
    int i = blockIdx.x * 256 + threadIdx.x;
    if (i >= total) return;
    const int wo = i % Wout;
    int t = i / Wout;
    const int ho = t % Hout;
    const int bc = t / Hout;
    float m = -3.402823466e38f;
    const unsigned short* plane = in + (size_t)bc * Hin * Win;
#pragma unroll
    for (int kh = 0; kh < 3; ++kh) {
#pragma unroll
        for (int kw = 0; kw < 3; ++kw) {
            const int ih = ho * 2 - 1 + kh;
            const int iw = wo * 2 - 1 + kw;
            if (ih >= 0 && ih < Hin && iw >= 0 && iw < Win) {
                const float v = bf2f(plane[ih * Win + iw]);
                m = (v > m) ? v : m;
            }
        }
    }
    out[i] = f2bf(m);
}

// ---------------------------------------------------------------------------
// Global average pool: [B,C,H,W] -> bf16 [B*C]
// ---------------------------------------------------------------------------
__global__ void avgpool_kernel(const unsigned short* __restrict__ in,
                               unsigned short* __restrict__ out,
                               int HW, int total) {
    int i = blockIdx.x * 256 + threadIdx.x;
    if (i >= total) return;
    const unsigned short* p = in + (size_t)i * HW;
    float s = 0.0f;
    for (int j = 0; j < HW; ++j) s += bf2f(p[j]);
    out[i] = f2bf(s / (float)HW);
}

// ---------------------------------------------------------------------------
// Host orchestration
// ---------------------------------------------------------------------------
static inline int rnd32(int k) { return (k + 31) & ~31; }

static void launch_conv(hipStream_t st,
                        const unsigned short* in, const unsigned short* wt,
                        const float* sc, const float* bi,
                        const unsigned short* res,
                        unsigned short* outb, float* outf,
                        int B, int Cin, int Hin, int Win,
                        int Cout, int KH, int KW, int stride, int pad,
                        int relu) {
    const int Hout = (Hin + 2 * pad - KH) / stride + 1;
    const int Wout = (Win + 2 * pad - KW) / stride + 1;
    const int K = Cin * KH * KW;
    const int Kpad = rnd32(K);
    const int totalPix = B * Hout * Wout;
    const int tiles = ((totalPix + 31) / 32) * ((Cout + 15) / 16);
    const int blocks = (tiles + 3) / 4;
    if (KH == 7) {
        conv_bn_wmma<7, 7><<<blocks, 128, 0, st>>>(in, wt, sc, bi, res, outb,
            outf, B, Cin, Hin, Win, Cout, Hout, Wout, stride, pad, Kpad, relu);
    } else if (KH == 3) {
        conv_bn_wmma<3, 3><<<blocks, 128, 0, st>>>(in, wt, sc, bi, res, outb,
            outf, B, Cin, Hin, Win, Cout, Hout, Wout, stride, pad, Kpad, relu);
    } else {
        conv_bn_wmma<1, 1><<<blocks, 128, 0, st>>>(in, wt, sc, bi, res, outb,
            outf, B, Cin, Hin, Win, Cout, Hout, Wout, stride, pad, Kpad, relu);
    }
}

static const int CFG[8][3] = {{64, 64, 1},  {64, 64, 1},  {64, 128, 2},
                              {128, 128, 1}, {128, 256, 2}, {256, 256, 1},
                              {256, 512, 2}, {512, 512, 1}};

extern "C" void kernel_launch(void* const* d_in, const int* in_sizes, int n_in,
                              void* d_out, int out_size, void* d_ws,
                              size_t ws_size, hipStream_t stream) {
    (void)in_sizes; (void)n_in; (void)out_size; (void)ws_size;
    const int B = 32;

    // ---- Parse inputs (setup_inputs dict insertion order, depth-first) ----
    int ii = 0;
    const float* x_f     = (const float*)d_in[ii++];
    const float* w_conv1 = (const float*)d_in[ii++];
    const float* s_bn1   = (const float*)d_in[ii++];
    const float* b_bn1   = (const float*)d_in[ii++];
    struct BP { const float *w1,*s1,*b1,*w2,*s2,*b2,*wd,*sd,*bd; int has_wd; };
    BP bp[8];
    for (int i = 0; i < 8; ++i) {
        const int cin = CFG[i][0], cout = CFG[i][1], s = CFG[i][2];
        bp[i].w1 = (const float*)d_in[ii++];
        bp[i].s1 = (const float*)d_in[ii++];
        bp[i].b1 = (const float*)d_in[ii++];
        bp[i].w2 = (const float*)d_in[ii++];
        bp[i].s2 = (const float*)d_in[ii++];
        bp[i].b2 = (const float*)d_in[ii++];
        bp[i].has_wd = (s != 1 || cin != cout);
        if (bp[i].has_wd) {
            bp[i].wd = (const float*)d_in[ii++];
            bp[i].sd = (const float*)d_in[ii++];
            bp[i].bd = (const float*)d_in[ii++];
        } else { bp[i].wd = nullptr; bp[i].sd = nullptr; bp[i].bd = nullptr; }
    }
    const float* w_fc = (const float*)d_in[ii++];
    const float* b_fc = (const float*)d_in[ii++];

    // ---- Carve workspace ----
    uintptr_t off = 0;
    char* base = (char*)d_ws;
    auto carve = [&](size_t bytes) -> void* {
        off = (off + 255) & ~(uintptr_t)255;
        void* p = base + off;
        off += bytes;
        return p;
    };
    const size_t MAXACT = (size_t)B * 64 * 112 * 112;   // largest activation
    unsigned short* Bf0   = (unsigned short*)carve(MAXACT * 2);
    unsigned short* Bf1   = (unsigned short*)carve(MAXACT * 2);
    unsigned short* Bf2   = (unsigned short*)carve(MAXACT * 2);
    unsigned short* wbase = (unsigned short*)carve((size_t)12 * 1024 * 1024 * 2);
    unsigned short* avgbf = (unsigned short*)carve((size_t)B * 512 * 2);

    // ---- Pack all weights fp32 -> bf16 [Cout][Kpad] ----
    unsigned short* wptr = wbase;
    auto pack_w = [&](const float* src, int Cout, int K) -> unsigned short* {
        const int Kpad = rnd32(K);
        unsigned short* dst = wptr;
        wptr += (size_t)Cout * Kpad;
        const int total = Cout * Kpad;
        pack_weight_bf16<<<(total + 255) / 256, 256, 0, stream>>>(src, dst,
                                                                  Cout, K, Kpad);
        return dst;
    };
    unsigned short* Wc1 = pack_w(w_conv1, 64, 3 * 49);
    unsigned short *Wb1[8], *Wb2[8], *Wbd[8];
    for (int i = 0; i < 8; ++i) {
        const int cin = CFG[i][0], cout = CFG[i][1];
        Wb1[i] = pack_w(bp[i].w1, cout, cin * 9);
        Wb2[i] = pack_w(bp[i].w2, cout, cout * 9);
        Wbd[i] = bp[i].has_wd ? pack_w(bp[i].wd, cout, cin) : nullptr;
    }
    unsigned short* Wfc = pack_w(w_fc, 1000, 512);

    // ---- Stem: x -> bf16, conv1 7x7 s2 + BN + ReLU, cut, maxpool ----
    const int xn = B * 3 * 224 * 224;
    pack_f32_bf16<<<(xn + 255) / 256, 256, 0, stream>>>(x_f, Bf1, xn);

    launch_conv(stream, Bf1, Wc1, s_bn1, b_bn1, nullptr, Bf0, nullptr,
                B, 3, 224, 224, 64, 7, 7, 2, 3, /*relu=*/1);
    cut_kernel<<<B * 64, 256, 0, stream>>>(Bf0, 112 * 112);
    {
        const int total = B * 64 * 56 * 56;
        maxpool_kernel<<<(total + 255) / 256, 256, 0, stream>>>(
            Bf0, Bf1, B * 64, 112, 112, 56, 56);
    }

    unsigned short* cur = Bf1;   // h
    unsigned short* tmp = Bf0;   // block intermediate
    unsigned short* idb = Bf2;   // identity / output rotate
    int H = 56;

    // ---- Residual blocks ----
    for (int i = 0; i < 8; ++i) {
        const int cin = CFG[i][0], cout = CFG[i][1], s = CFG[i][2];
        const int Hout = H / s;

        // conv1 3x3 (stride s) + BN1 + ReLU -> tmp ; cut
        launch_conv(stream, cur, Wb1[i], bp[i].s1, bp[i].b1, nullptr,
                    tmp, nullptr, B, cin, H, H, cout, 3, 3, s, 1, 1);
        cut_kernel<<<B * cout, 256, 0, stream>>>(tmp, Hout * Hout);

        // downsample identity (1x1 stride s + BNd) if needed
        const unsigned short* res;
        unsigned short* outp;
        if (bp[i].has_wd) {
            launch_conv(stream, cur, Wbd[i], bp[i].sd, bp[i].bd, nullptr,
                        idb, nullptr, B, cin, H, H, cout, 1, 1, s, 0, 0);
            res  = idb;
            outp = cur;              // old h is dead after conv1/convd read it
        } else {
            res  = cur;              // identity = h
            outp = idb;
        }

        // conv2 3x3 + BN2 + residual + ReLU -> outp ; cut
        launch_conv(stream, tmp, Wb2[i], bp[i].s2, bp[i].b2, res,
                    outp, nullptr, B, cout, Hout, Hout, cout, 3, 3, 1, 1, 1);
        cut_kernel<<<B * cout, 256, 0, stream>>>(outp, Hout * Hout);

        if (!bp[i].has_wd) {         // rotate: new h lives in idb
            unsigned short* t = cur; cur = idb; idb = t;
        }                            // has_wd: new h already in cur
        H = Hout;
    }

    // ---- Global avg pool -> bf16 [B,512] ----
    {
        const int total = B * 512;
        avgpool_kernel<<<(total + 255) / 256, 256, 0, stream>>>(cur, avgbf,
                                                                H * H, total);
    }

    // ---- FC as WMMA GEMM: [32,512] x [512,1000]^T + fc_b -> f32 d_out ----
    launch_conv(stream, avgbf, Wfc, nullptr, b_fc, nullptr,
                nullptr, (float*)d_out,
                B, 512, 1, 1, 1000, 1, 1, 1, 0, /*relu=*/0);
}
